// Design2VecBase_42545946034516
// MI455X (gfx1250) — matrix operand order repaired and checked by reading, not compiled, not run on hardware
//
#include <hip/hip_runtime.h>
#include <cstdint>
#include <cstddef>

// ---------------------------------------------------------------- types
typedef __attribute__((ext_vector_type(16))) __bf16 v16bf;
typedef __attribute__((ext_vector_type(8)))  __bf16 v8bf;
typedef __attribute__((ext_vector_type(8)))  float  v8f;

// ---------------------------------------------------------------- dims
#define GG      4
#define NN      2048
#define FF      144
#define FPAD    160
#define BB      32
#define HH      128
#define MTP     192
#define MHID    256
#define LAYERS  4
#define APAD    40   // padded LDS row stride (bf16 elems): 80 B = 20 banks -> conflict-free

// ---------------------------------------------------------------- helpers
static __device__ __forceinline__ __bf16 to_bf16(float f) { return (__bf16)f; }

// A-operand tile (16x32, bf16) from global. Lane l(0..15) row M=l holds
// K=[k0..k0+15]; lanes 16..31 K=[k0+16..k0+31] (matches load_tileB).
static __device__ __forceinline__ v16bf load_tileA(const __bf16* mat, int ld,
                                                   int m0, int k0, int lane) {
  int hi = lane >> 4;
  const __bf16* row = mat + (size_t)(m0 + (lane & 15)) * (size_t)ld;
  v8bf a0 = *(const v8bf*)(row + k0 + 16 * hi);
  v8bf a1 = *(const v8bf*)(row + k0 + 16 * hi + 8);
  v16bf r;
#pragma unroll
  for (int i = 0; i < 8; ++i) { r[i] = a0[i]; r[8 + i] = a1[i]; }
  return r;
}

// B-operand tile (32x16, bf16) from a TRANSPOSED (NxK) matrix: lane n(0..15)
// col N=n holds K=[k0..k0+15], lanes 16..31 K=[k0+16..k0+31]. One 32B load.
static __device__ __forceinline__ v16bf load_tileB(const __bf16* matT, int ld,
                                                   int n0, int k0, int lane) {
  int hi = lane >> 4;
  const __bf16* row = matT + (size_t)(n0 + (lane & 15)) * (size_t)ld;
  return *(const v16bf*)(row + k0 + 16 * hi);
}

static __device__ __forceinline__ v8f wmma_bf16(v16bf a, v16bf b, v8f c) {
  return __builtin_amdgcn_wmma_f32_16x16x32_bf16(false, a, false, b,
                                                 (short)0, c, false, false);
}

// ---------------------------------------------------------------- prep kernels
__global__ void k_conv_as(const float* __restrict__ src, __bf16* __restrict__ dst, int n) {
  int i = blockIdx.x * blockDim.x + threadIdx.x;
  int stride = gridDim.x * blockDim.x;
  for (; i < n; i += stride) dst[i] = to_bf16(src[i]);
}

__global__ void k_conv_xs(const float* __restrict__ xs, __bf16* __restrict__ dst) {
  int i = blockIdx.x * blockDim.x + threadIdx.x;
  if (i >= GG * NN * FPAD) return;
  int k = i % FPAD;
  int r = (i / FPAD) % NN;
  int g = i / (FPAD * NN);
  dst[i] = (k < FF) ? to_bf16(xs[((size_t)g * NN + r) * FF + k]) : to_bf16(0.0f);
}

__global__ void k_conv_w0t(const float* __restrict__ W0, __bf16* __restrict__ dst) {
  int i = blockIdx.x * blockDim.x + threadIdx.x;
  if (i >= HH * FPAD) return;
  int n = i / FPAD, k = i % FPAD;
  dst[i] = (k < FF) ? to_bf16(W0[k * HH + n]) : to_bf16(0.0f);
}

__global__ void k_conv_wt(const float* __restrict__ Ws, __bf16* __restrict__ dst) {
  int i = blockIdx.x * blockDim.x + threadIdx.x;
  if (i >= LAYERS * HH * HH) return;
  int l = i / (HH * HH);
  int rem = i % (HH * HH);
  int n = rem / HH, k = rem % HH;
  dst[i] = to_bf16(Ws[(size_t)l * HH * HH + k * HH + n]);
}

// ---------------------------------------------------------------- K1: x0 = relu(xs @ W0 + b0)
__global__ void k_gemm_x0(const __bf16* __restrict__ xs, const __bf16* __restrict__ w0t,
                          const float* __restrict__ b0,
                          __bf16* __restrict__ xbf, float* __restrict__ x0f) {
  int g = blockIdx.y;
  int m0 = blockIdx.x * 64;
  int wave = threadIdx.x >> 5;
  int lane = threadIdx.x & 31;
  int n0 = wave * 16;
  const __bf16* A = xs + (size_t)g * NN * FPAD;
  v8f acc[4];
#pragma unroll
  for (int t = 0; t < 4; ++t)
#pragma unroll
    for (int i = 0; i < 8; ++i) acc[t][i] = 0.0f;

  for (int k0 = 0; k0 < FPAD; k0 += 32) {
    v16bf b = load_tileB(w0t, FPAD, n0, k0, lane);
#pragma unroll
    for (int t = 0; t < 4; ++t) {
      v16bf a = load_tileA(A, FPAD, m0 + 16 * t, k0, lane);
      acc[t] = wmma_bf16(a, b, acc[t]);
    }
  }
  int h = n0 + (lane & 15);
  int hi = lane >> 4;
  float bias = b0[h];
#pragma unroll
  for (int t = 0; t < 4; ++t)
#pragma unroll
    for (int r = 0; r < 8; ++r) {
      int m = m0 + 16 * t + 8 * hi + r;
      float v = fmaxf(acc[t][r] + bias, 0.0f);
      size_t o = ((size_t)g * NN + m) * HH + h;
      x0f[o] = v;
      xbf[o] = to_bf16(v);
    }
}

// ---------------------------------------------------------------- K2: yT = (x @ W_i)^T
__global__ void k_gemm_xw(const __bf16* __restrict__ xbf, const __bf16* __restrict__ wt,
                          __bf16* __restrict__ yT) {
  int g = blockIdx.y;
  int m0 = blockIdx.x * 64;
  int wave = threadIdx.x >> 5;
  int lane = threadIdx.x & 31;
  int n0 = wave * 16;
  const __bf16* A = xbf + (size_t)g * NN * HH;
  v8f acc[4];
#pragma unroll
  for (int t = 0; t < 4; ++t)
#pragma unroll
    for (int i = 0; i < 8; ++i) acc[t][i] = 0.0f;

  for (int k0 = 0; k0 < HH; k0 += 32) {
    v16bf b = load_tileB(wt, HH, n0, k0, lane);
#pragma unroll
    for (int t = 0; t < 4; ++t) {
      v16bf a = load_tileA(A, HH, m0 + 16 * t, k0, lane);
      acc[t] = wmma_bf16(a, b, acc[t]);
    }
  }
  int hG = n0 + (lane & 15);
  int hi = lane >> 4;
  union Pk { __bf16 h[8]; uint4 u; } pk;
#pragma unroll
  for (int t = 0; t < 4; ++t) {
#pragma unroll
    for (int r = 0; r < 8; ++r) pk.h[r] = to_bf16(acc[t][r]);
    size_t o = ((size_t)g * HH + hG) * NN + m0 + 16 * t + 8 * hi;
    *(uint4*)(yT + o) = pk.u;   // one column-slice per lane, contiguous in nodes
  }
}

// ---------------------------------------------------------------- K3: x = act(As @ y + b_i)
// Heavy kernel. A slab (64x32 bf16, shared by all 8 waves) is staged into LDS
// with double-buffered global_load_async_to_lds_b128 (ASYNCcnt); B is
// register-double-buffered from L2-resident yT.
// last==0: relu -> bf16 x ; last==1: softmax(H) + residual -> f32 final
__global__ void k_gemm_agg(const __bf16* __restrict__ asbf, const __bf16* __restrict__ yT,
                           const float* __restrict__ bias, const float* __restrict__ x0f,
                           __bf16* __restrict__ xbf, float* __restrict__ finalf, int last) {
  __shared__ float sm[64 * HH];                          // 32 KB softmax slab
  __shared__ __align__(16) __bf16 smA[2][64 * APAD];     // 2 x 5 KB A tiles
  int g = blockIdx.y;
  int m0 = blockIdx.x * 64;
  int tid = threadIdx.x;
  int wave = tid >> 5;
  int lane = tid & 31;
  int n0 = wave * 16;
  int hi = lane >> 4;

  // staging map: thread tid moves one 16-B chunk: row = tid/4, chunk = tid%4
  int srow = tid >> 2, schunk = tid & 3;
  const __bf16* gA = asbf + (size_t)g * NN * NN + (size_t)(m0 + srow) * NN + schunk * 8;
  unsigned ldsA0 = (unsigned)(size_t)(&smA[0][srow * APAD + schunk * 8]);
  unsigned ldsA1 = (unsigned)(size_t)(&smA[1][srow * APAD + schunk * 8]);

  // per-lane B stream base (transposed yT: row = output col h, contiguous in k)
  const __bf16* Bt = yT + (size_t)g * HH * NN
                   + (size_t)(n0 + (lane & 15)) * NN + 16 * hi;

  v8f acc[4];
#pragma unroll
  for (int t = 0; t < 4; ++t)
#pragma unroll
    for (int i = 0; i < 8; ++i) acc[t][i] = 0.0f;

  // prologue: stage step 0 into buffer 0; fetch first B tile
  asm volatile("global_load_async_to_lds_b128 %0, %1, off"
               :: "v"(ldsA0), "v"(gA) : "memory");
  v16bf bcur = *(const v16bf*)Bt;

  const int KSTEPS = NN / 32;
  for (int step = 0; step < KSTEPS; ++step) {
    int cur = step & 1;
    if (step + 1 < KSTEPS) {
      // kick off async DMA for the next A slab into the other buffer
      asm volatile("global_load_async_to_lds_b128 %0, %1, off"
                   :: "v"(cur ? ldsA0 : ldsA1),
                      "v"(gA + (size_t)(step + 1) * 32) : "memory");
      asm volatile("s_wait_asynccnt 0x1" ::: "memory");   // current slab landed
    } else {
      asm volatile("s_wait_asynccnt 0x0" ::: "memory");
    }
    __syncthreads();                                      // slab visible to all waves

    v16bf bnext = bcur;
    if (step + 1 < KSTEPS) {
      bnext = *(const v16bf*)(Bt + (size_t)(step + 1) * 32);
      if (step + 2 < KSTEPS)
        __builtin_prefetch(Bt + (size_t)(step + 2) * 32, 0, 1);
    }

    const __bf16* At = &smA[cur][(lane & 15) * APAD];
#pragma unroll
    for (int t = 0; t < 4; ++t) {
      const __bf16* rp = At + t * 16 * APAD;
      v8bf a0 = *(const v8bf*)(rp + 16 * hi);             // ds_load_b128
      v8bf a1 = *(const v8bf*)(rp + 16 * hi + 8);         // ds_load_b128
      v16bf a;
#pragma unroll
      for (int i = 0; i < 8; ++i) { a[i] = a0[i]; a[8 + i] = a1[i]; }
      acc[t] = wmma_bf16(a, bcur, acc[t]);
    }
    bcur = bnext;
    __syncthreads();                    // reads done before buffer is overwritten
  }

  int h = n0 + (lane & 15);
  float bb = bias[h];
  if (!last) {
#pragma unroll
    for (int t = 0; t < 4; ++t)
#pragma unroll
      for (int r = 0; r < 8; ++r) {
        int m = m0 + 16 * t + 8 * hi + r;
        xbf[((size_t)g * NN + m) * HH + h] = to_bf16(fmaxf(acc[t][r] + bb, 0.0f));
      }
  } else {
#pragma unroll
    for (int t = 0; t < 4; ++t)
#pragma unroll
      for (int r = 0; r < 8; ++r)
        sm[(16 * t + 8 * hi + r) * HH + h] = acc[t][r] + bb;
    __syncthreads();
    if (tid < 64) {
      const float* rowp = sm + tid * HH;
      float mx = rowp[0];
      for (int j = 1; j < HH; ++j) mx = fmaxf(mx, rowp[j]);
      float s = 0.0f;
      for (int j = 0; j < HH; ++j) s += __expf(rowp[j] - mx);
      float inv = 1.0f / s;
      size_t base = ((size_t)g * NN + (m0 + tid)) * HH;
      for (int j = 0; j < HH; ++j)
        finalf[base + j] = __expf(rowp[j] - mx) * inv + x0f[base + j];
    }
  }
}

// ---------------------------------------------------------------- K4: masked mean per batch
__global__ void k_cov(const float* __restrict__ finalf, const int* __restrict__ gidx,
                      const unsigned char* __restrict__ masks, float* __restrict__ cov) {
  int b = blockIdx.x;
  int h = threadIdx.x;  // 128 threads
  int g = gidx[b];
  const float* xp = finalf + (size_t)g * NN * HH + h;
  const unsigned char* mp = masks + (size_t)b * NN;
  float s = 0.0f, c = 0.0f;
  for (int n = 0; n < NN; ++n)
    if (mp[n]) { s += xp[(size_t)n * HH]; c += 1.0f; }
  cov[b * HH + h] = s / fmaxf(c, 1.0f);
}

// ---------------------------------------------------------------- K5: tp MLP (tiny)
__global__ void k_tp(const float* __restrict__ tps, const float* __restrict__ W1,
                     const float* __restrict__ b1, const float* __restrict__ W2,
                     const float* __restrict__ b2, float* __restrict__ tpout) {
  __shared__ float st[MTP];
  __shared__ float z1[MHID];
  __shared__ float ev[MHID];
  int b = blockIdx.x;
  int t = threadIdx.x;  // 256
  if (t < MTP) st[t] = tps[b * MTP + t];
  __syncthreads();
  float a = b1[t];
  for (int k = 0; k < MTP; ++k) a += st[k] * W1[k * MHID + t];
  z1[t] = fmaxf(a, 0.0f);
  __syncthreads();
  float a2 = b2[t];
  for (int k = 0; k < MHID; ++k) a2 += z1[k] * W2[k * MHID + t];
  ev[t] = a2;
  __syncthreads();
  float mx = ev[0];
  for (int k = 1; k < MHID; ++k) mx = fmaxf(mx, ev[k]);
  float ex = __expf(a2 - mx);
  __syncthreads();
  ev[t] = ex;
  __syncthreads();
  float s = 0.0f;
  for (int k = 0; k < MHID; ++k) s += ev[k];
  tpout[b * MHID + t] = ex / s;
}

// ---------------------------------------------------------------- K6: head
__global__ void k_head(const float* __restrict__ cov, const float* __restrict__ tpout,
                       const float* __restrict__ Wf1, const float* __restrict__ bf1,
                       const float* __restrict__ Wf2, const float* __restrict__ bf2,
                       float* __restrict__ out) {
  __shared__ float z[HH + MHID];  // 384
  __shared__ float h1[MHID];
  __shared__ float red[MHID];
  int b = blockIdx.x;
  int t = threadIdx.x;  // 256
  if (t < HH) z[t] = cov[b * HH + t];
  z[HH + t] = tpout[b * MHID + t];
  __syncthreads();
  float a = bf1[t];
  for (int k = 0; k < HH + MHID; ++k) a += z[k] * Wf1[k * MHID + t];
  h1[t] = fmaxf(a, 0.0f);
  __syncthreads();
  red[t] = h1[t] * Wf2[t];
  __syncthreads();
  for (int s = 128; s > 0; s >>= 1) {
    if (t < s) red[t] += red[t + s];
    __syncthreads();
  }
  if (t == 0) out[b] = 1.0f / (1.0f + __expf(-(red[0] + bf2[0])));
}

// ---------------------------------------------------------------- launch
extern "C" void kernel_launch(void* const* d_in, const int* in_sizes, int n_in,
                              void* d_out, int out_size, void* d_ws, size_t ws_size,
                              hipStream_t stream) {
  const float* graph_xs = (const float*)d_in[0];
  const float* graph_as = (const float*)d_in[1];
  const int* gidx       = (const int*)d_in[2];
  const unsigned char* masks = (const unsigned char*)d_in[3];
  const float* tps  = (const float*)d_in[4];
  const float* W0   = (const float*)d_in[5];
  const float* b0   = (const float*)d_in[6];
  const float* gWs  = (const float*)d_in[7];
  const float* gbs  = (const float*)d_in[8];
  const float* Wtp1 = (const float*)d_in[9];
  const float* btp1 = (const float*)d_in[10];
  const float* Wtp2 = (const float*)d_in[11];
  const float* btp2 = (const float*)d_in[12];
  const float* Wf1  = (const float*)d_in[13];
  const float* bf1  = (const float*)d_in[14];
  const float* Wf2  = (const float*)d_in[15];
  const float* bf2  = (const float*)d_in[16];
  float* out = (float*)d_out;

  // workspace layout (all offsets multiples of 4 KB)
  char* ws = (char*)d_ws;
  size_t off = 0;
  __bf16* As_bf = (__bf16*)(ws + off); off += (size_t)GG * NN * NN * 2;       // 32 MB
  __bf16* xs_bf = (__bf16*)(ws + off); off += (size_t)GG * NN * FPAD * 2;     // 2.5 MB
  __bf16* W0t   = (__bf16*)(ws + off); off += (size_t)HH * FPAD * 2;          // 40 KB
  __bf16* Wt    = (__bf16*)(ws + off); off += (size_t)LAYERS * HH * HH * 2;   // 128 KB
  __bf16* x_bf  = (__bf16*)(ws + off); off += (size_t)GG * NN * HH * 2;       // 2 MB
  float*  x0f   = (float*)(ws + off);  off += (size_t)GG * NN * HH * 4;       // 4 MB
  __bf16* yT    = (__bf16*)(ws + off); off += (size_t)GG * HH * NN * 2;       // 2 MB
  float*  finf  = (float*)(ws + off);  off += (size_t)GG * NN * HH * 4;       // 4 MB
  float*  cov   = (float*)(ws + off);  off += (size_t)BB * HH * 4;
  float*  tpo   = (float*)(ws + off);  off += (size_t)BB * MHID * 4;
  (void)ws_size; (void)in_sizes; (void)n_in; (void)out_size;

  // precision conversion / transposition
  k_conv_as<<<8192, 256, 0, stream>>>(graph_as, As_bf, GG * NN * NN);
  k_conv_xs<<<(GG * NN * FPAD + 255) / 256, 256, 0, stream>>>(graph_xs, xs_bf);
  k_conv_w0t<<<(HH * FPAD + 255) / 256, 256, 0, stream>>>(W0, W0t);
  k_conv_wt<<<(LAYERS * HH * HH + 255) / 256, 256, 0, stream>>>(gWs, Wt);

  dim3 grid(NN / 64, GG), blk(256);
  k_gemm_x0<<<grid, blk, 0, stream>>>(xs_bf, W0t, b0, x_bf, x0f);

  for (int i = 0; i < LAYERS; ++i) {
    k_gemm_xw<<<grid, blk, 0, stream>>>(x_bf, Wt + (size_t)i * HH * HH, yT);
    k_gemm_agg<<<grid, blk, 0, stream>>>(As_bf, yT, gbs + (size_t)i * HH, x0f,
                                         x_bf, finf, (i == LAYERS - 1) ? 1 : 0);
  }

  k_cov<<<BB, HH, 0, stream>>>(finf, gidx, masks, cov);
  k_tp<<<BB, 256, 0, stream>>>(tps, Wtp1, btp1, Wtp2, btp2, tpo);
  k_head<<<BB, 256, 0, stream>>>(cov, tpo, Wf1, bf1, Wf2, bf2, out);
}
